// GlobalLocalAttention_20873541058774
// MI455X (gfx1250) — matrix operand order repaired and checked by reading, not compiled, not run on hardware
//
#include <hip/hip_runtime.h>
#include <hip/hip_bf16.h>
#include <math.h>

typedef _Float16 f16;
typedef _Float16 v16h __attribute__((ext_vector_type(16)));
typedef float    v8f  __attribute__((ext_vector_type(8)));
typedef float    v4f  __attribute__((ext_vector_type(4)));
typedef int      v4i  __attribute__((ext_vector_type(4)));

#define HWD   64
#define NPIX  4096
#define CCH   64
#define LP    36864      /* L * 9 */
#define KCR   576        /* C * 9 */
#define EPSV  1e-7f

// ---------------------------------------------------------------------------
// WMMA helpers (CDNA5 wave32, v_wmma_f32_16x16x32_f16)
// ---------------------------------------------------------------------------
__device__ __forceinline__ v8f wmma_f16(v8f acc, v16h a, v16h b) {
  return __builtin_amdgcn_wmma_f32_16x16x32_f16(false, a, false, b,
                                                (short)0, acc, false, false);
}

union U16 { v16h h; struct { v4i lo, hi; } q; };

// A fragment 16x32 at (m0,k0), row-major A[lda] f16: two b128 loads.
// lane<16: row=m0+lane, K {k0..k0+15}; lane>=16: row-16, +16 K offset
__device__ __forceinline__ v16h load_a16v(const f16* A, size_t lda, int m0, int k0) {
  int lane = threadIdx.x & 31;
  const f16* p = A + (size_t)(m0 + (lane & 15)) * lda + (k0 + ((lane >> 4) << 4));
  U16 u;
  u.q.lo = *(const v4i*)p;         // K + 0..7
  u.q.hi = *(const v4i*)(p + 8);   // K + 8..15
  return u.h;
}

// B fragment 32x16 from swizzled panels Bs[k>>5][n][k&31] (nld = N*32): two b128s.
__device__ __forceinline__ v16h load_bs(const f16* Bs, size_t nld, int k0, int n0) {
  int lane = threadIdx.x & 31;
  const f16* p = Bs + (size_t)(k0 >> 5) * nld + ((size_t)(n0 + (lane & 15)) << 5)
               + ((lane >> 4) << 4);
  U16 u;
  u.q.lo = *(const v4i*)p;
  u.q.hi = *(const v4i*)(p + 8);
  return u.h;
}

__device__ __forceinline__ size_t swb(int k, int n, size_t nld) {
  return (size_t)(k >> 5) * nld + ((size_t)n << 5) + (k & 31);
}

// ---------------------------------------------------------------------------
// Prep
// ---------------------------------------------------------------------------
__global__ void k_prep_weights(const float* gq, const float* gk, const float* gv,
                               const float* pq, const float* pv,
                               const float* pk, const float* pbk,
                               f16* w16, float* wks) {
  int tid = blockIdx.x * 256 + threadIdx.x;
  const float* srcs[5] = {gq, gk, gv, pq, pv};
  if (tid < 5 * 4096) w16[tid] = (f16)srcs[tid >> 12][tid & 4095];
  if (tid < 64) {
    float s = 0.f;
    for (int d = 0; d < 64; ++d) s += pk[d * 64 + tid];
    wks[tid] = s;
  }
  if (tid == 64) {
    float s = 0.f;
    for (int d = 0; d < 64; ++d) s += pbk[d];
    wks[64] = s;
  }
}

// a16s / bg16s in swizzled-B panels, K=channel (2 chunks), N=pixel
__global__ void k_pack_fg(const float* fg, const float* mask, f16* a16s, f16* bg16s) {
  int i = blockIdx.x * 256 + threadIdx.x;   // 0 .. C*N-1
  int c = i >> 12, n = i & (NPIX - 1);
  float m = mask[n];
  float v = fg[i];
  size_t d = swb(c, n, (size_t)NPIX * 32);
  a16s[d]  = (f16)v;
  bg16s[d] = (f16)(v * (1.f - m));
}

// ---------------------------------------------------------------------------
// Global attention
// ---------------------------------------------------------------------------
// 1x1 conv: D = W(64x64)*B(64x4096)+bias.
// MODE 0: store swizzled-B panels, K=channel (k16s)
// MODE 1: store row-major transposed qT[n][64] (A operand of sim)
// MODE 2: store swizzled-B panels, K=pixel, N=channel (vTs, nld=2048)
template <int MODE>
__global__ void k_gemm_qkv(const f16* W, const f16* Bs, const float* bias, f16* out) {
  int n0 = blockIdx.x * 16, m0 = blockIdx.y * 16;
  v8f acc = {};
  for (int k0 = 0; k0 < 64; k0 += 32)
    acc = wmma_f16(acc, load_a16v(W, 64, m0, k0),
                   load_bs(Bs, (size_t)NPIX * 32, k0, n0));
  int lane = threadIdx.x;
  int n = n0 + (lane & 15), mb = m0 + ((lane >> 4) << 3);
  if (MODE == 2) {
#pragma unroll
    for (int r = 0; r < 8; ++r)
      out[swb(n, mb + r, 2048)] = (f16)(acc[r] + bias[mb + r]);
  } else {
    union { v4i v; f16 e[8]; } u;
#pragma unroll
    for (int r = 0; r < 8; ++r) u.e[r] = (f16)(acc[r] + bias[mb + r]);
    if (MODE == 1) *(v4i*)(out + (size_t)n * 64 + mb) = u.v;
    else *(v4i*)(out + (size_t)(mb >> 5) * ((size_t)NPIX * 32)
                 + ((size_t)n << 5) + (mb & 31)) = u.v;
  }
}

// sim[i][j] = (qT[i][:] . k[:][j]) * mask[i]   (32x32 tile / wave)
__global__ void k_gemm_sim(const f16* qT, const f16* kS, const float* mask, float* sim) {
  int n0 = blockIdx.x * 32, m0 = blockIdx.y * 32;
  v8f a00 = {}, a01 = {}, a10 = {}, a11 = {};
  for (int k0 = 0; k0 < 64; k0 += 32) {
    v16h A0 = load_a16v(qT, 64, m0, k0), A1 = load_a16v(qT, 64, m0 + 16, k0);
    v16h B0 = load_bs(kS, (size_t)NPIX * 32, k0, n0);
    v16h B1 = load_bs(kS, (size_t)NPIX * 32, k0, n0 + 16);
    a00 = wmma_f16(a00, A0, B0); a01 = wmma_f16(a01, A0, B1);
    a10 = wmma_f16(a10, A1, B0); a11 = wmma_f16(a11, A1, B1);
  }
  int lane = threadIdx.x, nn = lane & 15, mo = (lane >> 4) << 3;
#pragma unroll
  for (int r = 0; r < 8; ++r) {
    int ma = m0 + mo + r, mb = m0 + 16 + mo + r;
    float fa = mask[ma], fb = mask[mb];
    sim[(size_t)ma * NPIX + n0 + nn]      = a00[r] * fa;
    sim[(size_t)ma * NPIX + n0 + 16 + nn] = a01[r] * fa;
    sim[(size_t)mb * NPIX + n0 + nn]      = a10[r] * fb;
    sim[(size_t)mb * NPIX + n0 + 16 + nn] = a11[r] * fb;
  }
}

__global__ void k_softmax_row(const float* sim, f16* attn) {
  int row = blockIdx.x, tid = threadIdx.x;
  const float* p = sim + (size_t)row * NPIX;
  __shared__ float red[256];
  float mx = -1e30f;
  for (int j = tid; j < NPIX; j += 256) mx = fmaxf(mx, p[j]);
  red[tid] = mx; __syncthreads();
  for (int o = 128; o > 0; o >>= 1) {
    if (tid < o) red[tid] = fmaxf(red[tid], red[tid + o]);
    __syncthreads();
  }
  mx = red[0]; __syncthreads();
  float s = 0.f;
  for (int j = tid; j < NPIX; j += 256) s += __expf(p[j] - mx);
  red[tid] = s; __syncthreads();
  for (int o = 128; o > 0; o >>= 1) {
    if (tid < o) red[tid] += red[tid + o];
    __syncthreads();
  }
  float inv = 1.f / red[0];
  f16* q = attn + (size_t)row * NPIX;
  for (int j = tid; j < NPIX; j += 256) q[j] = (f16)(__expf(p[j] - mx) * inv);
}

// outT[i][c] = sum_j attn[i][j]*vT[j][c]; fgout[c][i] = fg*m + g*(1-m)*outT
__global__ void k_gemm_out(const f16* attn, const f16* vTs, const float* fg,
                           const float* mask, const float* gamma, float* fgout) {
  int n0 = blockIdx.x * 32, m0 = blockIdx.y * 32;   // n: channel, m: pixel
  v8f a00 = {}, a01 = {}, a10 = {}, a11 = {};
  for (int k0 = 0; k0 < NPIX; k0 += 32) {
    v16h A0 = load_a16v(attn, NPIX, m0, k0), A1 = load_a16v(attn, NPIX, m0 + 16, k0);
    v16h B0 = load_bs(vTs, 2048, k0, n0), B1 = load_bs(vTs, 2048, k0, n0 + 16);
    a00 = wmma_f16(a00, A0, B0); a01 = wmma_f16(a01, A0, B1);
    a10 = wmma_f16(a10, A1, B0); a11 = wmma_f16(a11, A1, B1);
  }
  int lane = threadIdx.x;
  int c0 = n0 + (lane & 15), c1 = c0 + 16;
  int i0 = m0 + ((lane >> 4) << 3), i1 = i0 + 16;
  float g = gamma[0];
  v4f o[8];
#pragma unroll
  for (int r = 0; r < 8; ++r) {
    int ia = i0 + r, ib = i1 + r;
    float ma = mask[ia], mb = mask[ib];
    o[(r >> 2) + 0][r & 3] = fg[(size_t)c0 * NPIX + ia] * ma + g * (1.f - ma) * a00[r];
    o[(r >> 2) + 2][r & 3] = fg[(size_t)c1 * NPIX + ia] * ma + g * (1.f - ma) * a01[r];
    o[(r >> 2) + 4][r & 3] = fg[(size_t)c0 * NPIX + ib] * mb + g * (1.f - mb) * a10[r];
    o[(r >> 2) + 6][r & 3] = fg[(size_t)c1 * NPIX + ib] * mb + g * (1.f - mb) * a11[r];
  }
  *(v4f*)(fgout + (size_t)c0 * NPIX + i0)     = o[0];
  *(v4f*)(fgout + (size_t)c0 * NPIX + i0 + 4) = o[1];
  *(v4f*)(fgout + (size_t)c1 * NPIX + i0)     = o[2];
  *(v4f*)(fgout + (size_t)c1 * NPIX + i0 + 4) = o[3];
  *(v4f*)(fgout + (size_t)c0 * NPIX + i1)     = o[4];
  *(v4f*)(fgout + (size_t)c0 * NPIX + i1 + 4) = o[5];
  *(v4f*)(fgout + (size_t)c1 * NPIX + i1)     = o[6];
  *(v4f*)(fgout + (size_t)c1 * NPIX + i1 + 4) = o[7];
}

// ---------------------------------------------------------------------------
// Patch attention
// ---------------------------------------------------------------------------
// xu swizzled panels: K=channel (2 chunks), N=lp (L*9). value = unfold(bg)+EPS
__global__ void k_pack_xu(const float* fg, const float* mask, f16* xus) {
  int idx = blockIdx.x * 256 + threadIdx.x;     // 0 .. 64*LP-1
  int c = idx & 63, lp = idx >> 6;
  int l = lp / 9, t = lp - 9 * l;
  int y = l >> 6, x = l & 63;
  int sy = y + t / 3 - 1, sx = x + (t % 3) - 1;
  float v = 0.f;
  if ((unsigned)sy < 64u && (unsigned)sx < 64u) {
    int pix = (sy << 6) + sx;
    v = fg[((size_t)c << 12) + pix] * (1.f - mask[pix]);
  }
  xus[swb(c, lp, (size_t)LP * 32)] = (f16)(v + EPSV);
}

__global__ void k_pack_mu(const float* mask, float* mu) {
  int lp = blockIdx.x * 256 + threadIdx.x;      // 0 .. LP-1
  int l = lp / 9, t = lp - 9 * l;
  int y = l >> 6, x = l & 63;
  int sy = y + t / 3 - 1, sx = x + (t % 3) - 1;
  float v = 0.f;
  if ((unsigned)sy < 64u && (unsigned)sx < 64u) v = mask[(sy << 6) + sx];
  mu[lp] = v;
}

// patch q/v projection: outT[lp][c] = (W*xu)[c][lp] + bias[c]  (f32)
__global__ void k_gemm_pa(const f16* W, const f16* xus, const float* bias, float* outT) {
  int n0 = blockIdx.x * 16, m0 = blockIdx.y * 16;
  v8f acc = {};
  for (int k0 = 0; k0 < 64; k0 += 32)
    acc = wmma_f16(acc, load_a16v(W, 64, m0, k0),
                   load_bs(xus, (size_t)LP * 32, k0, n0));
  int lane = threadIdx.x;
  int n = n0 + (lane & 15), mb = m0 + ((lane >> 4) << 3);
  v4f o0, o1;
#pragma unroll
  for (int r = 0; r < 4; ++r) {
    o0[r] = acc[r] + bias[mb + r];
    o1[r] = acc[r + 4] + bias[mb + 4 + r];
  }
  *(v4f*)(outT + (size_t)n * 64 + mb) = o0;
  *(v4f*)(outT + (size_t)n * 64 + mb + 4) = o1;
}

// ksum[lp] = sum_c wks[c]*x[c][lp] + wks[64]
__global__ void k_ksum(const f16* xus, const float* wks, float* ksum) {
  int lp = blockIdx.x * 256 + threadIdx.x;
  float s = wks[64];
  for (int c = 0; c < 64; ++c)
    s += wks[c] * (float)xus[(size_t)(c >> 5) * ((size_t)LP * 32)
                             + ((size_t)lp << 5) + (c & 31)];
  ksum[lp] = s;
}

// Per-patch attention + L2 normalization. Block = 64 threads (c), grid = L.
__global__ void k_patch(const f16* xus, const float* qT, const float* vT,
                        const float* ksum, const float* mu, const float* gamma,
                        f16* ckA_cr, f16* ckA_rec, float* mm) {
  int l = blockIdx.x, c = threadIdx.x;
  __shared__ float red[64];
  float muv[9], ks[9];
#pragma unroll
  for (int p = 0; p < 9; ++p) { muv[p] = mu[l * 9 + p]; ks[p] = ksum[l * 9 + p]; }
  float s = 0.f;
#pragma unroll
  for (int p = 0; p < 9; ++p) s += qT[(size_t)(l * 9 + p) * 64 + c] * ks[p];
  float e[9], mx = -1e30f;
#pragma unroll
  for (int p = 0; p < 9; ++p) { e[p] = s * muv[p]; mx = fmaxf(mx, e[p]); }
  float sum = 0.f;
#pragma unroll
  for (int p = 0; p < 9; ++p) { e[p] = __expf(e[p] - mx); sum += e[p]; }
  float inv = 1.f / sum, g = gamma[0];
  float ck[9], ss = 0.f;
#pragma unroll
  for (int p = 0; p < 9; ++p) {
    float att = e[p] * inv * vT[(size_t)(l * 9 + p) * 64 + c];
    float xv  = (float)xus[(size_t)(c >> 5) * ((size_t)LP * 32)
                           + ((size_t)(l * 9 + p) << 5) + (c & 31)];
    float v   = xv * muv[p] + g * (1.f - muv[p]) * att;
    ck[p] = v; ss += v * v;
  }
  red[c] = ss; __syncthreads();
  for (int o = 32; o > 0; o >>= 1) {
    if (c < o) red[c] += red[c + o];
    __syncthreads();
  }
  float inorm = 1.f / sqrtf(red[0]);
#pragma unroll
  for (int p = 0; p < 9; ++p) {
    f16 h = (f16)(ck[p] * inorm);
    ckA_cr[(size_t)l * KCR + c * 9 + p] = h;                 // [L][C*9]
    ckA_rec[(size_t)c * LP + (8 - p) * NPIX + l] = h;        // [C][9*L], flipped
  }
  if (c == 0) {
    float mn = 0.f;
#pragma unroll
    for (int p = 0; p < 9; ++p) mn += muv[p];
    mm[l] = (mn * (1.f / 9.f) == 0.f) ? 1.f : 0.f;
  }
}

// Build swizzled panels of the unfolded global-attended map: K=c*9+t, N=pixel
__global__ void k_pack_bcr(const float* fgout, f16* Bcrs) {
  int idx = blockIdx.x * 256 + threadIdx.x;   // 0 .. 576*4096-1
  int n = idx / KCR, k = idx - n * KCR;
  int c = k / 9, t = k - 9 * c;
  int y = n >> 6, x = n & 63;
  int sy = y + t / 3 - 1, sx = x + (t % 3) - 1;
  float v = 0.f;
  if ((unsigned)sy < 64u && (unsigned)sx < 64u)
    v = fgout[((size_t)c << 12) + (sy << 6) + sx];
  Bcrs[swb(k, n, (size_t)NPIX * 32)] = (f16)v;
}

// cr = ckA_cr(4096x576) * Bcrs(576x4096)   (32x32 tile / wave)
__global__ void k_gemm_cr(const f16* ckA, const f16* Bcrs, float* cr) {
  int n0 = blockIdx.x * 32, m0 = blockIdx.y * 32;
  v8f a00 = {}, a01 = {}, a10 = {}, a11 = {};
  for (int k0 = 0; k0 < KCR; k0 += 32) {
    v16h A0 = load_a16v(ckA, KCR, m0, k0), A1 = load_a16v(ckA, KCR, m0 + 16, k0);
    v16h B0 = load_bs(Bcrs, (size_t)NPIX * 32, k0, n0);
    v16h B1 = load_bs(Bcrs, (size_t)NPIX * 32, k0, n0 + 16);
    a00 = wmma_f16(a00, A0, B0); a01 = wmma_f16(a01, A0, B1);
    a10 = wmma_f16(a10, A1, B0); a11 = wmma_f16(a11, A1, B1);
  }
  int lane = threadIdx.x, nn = lane & 15, mo = (lane >> 4) << 3;
#pragma unroll
  for (int r = 0; r < 8; ++r) {
    int ma = m0 + mo + r, mb = m0 + 16 + mo + r;
    cr[(size_t)ma * NPIX + n0 + nn]      = a00[r];
    cr[(size_t)ma * NPIX + n0 + 16 + nn] = a01[r];
    cr[(size_t)mb * NPIX + n0 + nn]      = a10[r];
    cr[(size_t)mb * NPIX + n0 + 16 + nn] = a11[r];
  }
}

// 3x3 box filter per channel l (zero pad)
__global__ void k_box(const float* cr, float* cr2) {
  __shared__ float img[NPIX];
  int l = blockIdx.x, tid = threadIdx.x;
  const float* src = cr + (size_t)l * NPIX;
  for (int i = tid; i < NPIX; i += 256) img[i] = src[i];
  __syncthreads();
  float* dst = cr2 + (size_t)l * NPIX;
  for (int i = tid; i < NPIX; i += 256) {
    int y = i >> 6, x = i & 63;
    float s = 0.f;
    for (int dy = -1; dy <= 1; ++dy) {
      int sy = y + dy; if ((unsigned)sy >= 64u) continue;
      for (int dx = -1; dx <= 1; ++dx) {
        int sx = x + dx; if ((unsigned)sx >= 64u) continue;
        s += img[(sy << 6) + sx];
      }
    }
    dst[i] = s;
  }
}

// column softmax over l; writes attn2 in swizzled panels K=l, N=pixel
__global__ void k_colsoftmax(const float* cr2, const float* mm, f16* attn2s) {
  int tx = threadIdx.x & 63, rp = threadIdx.x >> 6;
  int col = blockIdx.x * 64 + tx;
  __shared__ float rbuf[4][64];
  float mx = -1e30f;
  for (int l = rp; l < NPIX; l += 4)
    mx = fmaxf(mx, cr2[(size_t)l * NPIX + col] * mm[l]);
  rbuf[rp][tx] = mx; __syncthreads();
  mx = fmaxf(fmaxf(rbuf[0][tx], rbuf[1][tx]), fmaxf(rbuf[2][tx], rbuf[3][tx]));
  __syncthreads();
  float s = 0.f;
  for (int l = rp; l < NPIX; l += 4)
    s += __expf(cr2[(size_t)l * NPIX + col] * mm[l] - mx);
  rbuf[rp][tx] = s; __syncthreads();
  s = rbuf[0][tx] + rbuf[1][tx] + rbuf[2][tx] + rbuf[3][tx];
  float inv = 1.f / s;
  for (int l = rp; l < NPIX; l += 4) {
    float v = __expf(cr2[(size_t)l * NPIX + col] * mm[l] - mx) * inv * mm[l];
    attn2s[swb(l, col, (size_t)NPIX * 32)] = (f16)v;
  }
}

// rec[c][n] = sum_{t,l} ckA_rec[c][t*4096+l] * attn2[l, shift_t(n)]
// B fragments: per-tap base + spatial shift, zero fragment at borders.
__global__ void k_gemm_rec(const f16* ckA, const f16* attn2s, float* out) {
  int n0 = blockIdx.x * 32, m0 = blockIdx.y * 32;
  int lane = threadIdx.x;
  int c0 = n0 + (lane & 15), c1 = c0 + 16;
  int y0 = c0 >> 6, x0 = c0 & 63, y1 = c1 >> 6, x1 = c1 & 63;
  int kh16 = (lane >> 4) << 4;
  v8f a00 = {}, a01 = {}, a10 = {}, a11 = {};
  for (int k0 = 0; k0 < LP; k0 += 32) {
    v16h A0 = load_a16v(ckA, LP, m0, k0), A1 = load_a16v(ckA, LP, m0 + 16, k0);
    int t = k0 >> 12;                         // uniform over fragment
    int dy = t / 3 - 1, dx = (t % 3) - 1;
    const f16* cb = attn2s + (size_t)((k0 & 4095) >> 5) * ((size_t)NPIX * 32) + kh16;
    v16h B0 = {}, B1 = {};
    int sy = y0 + dy, sx = x0 + dx;
    if ((unsigned)sy < 64u && (unsigned)sx < 64u) {
      const f16* p = cb + ((size_t)((sy << 6) + sx) << 5);
      U16 u; u.q.lo = *(const v4i*)p; u.q.hi = *(const v4i*)(p + 8); B0 = u.h;
    }
    sy = y1 + dy; sx = x1 + dx;
    if ((unsigned)sy < 64u && (unsigned)sx < 64u) {
      const f16* p = cb + ((size_t)((sy << 6) + sx) << 5);
      U16 u; u.q.lo = *(const v4i*)p; u.q.hi = *(const v4i*)(p + 8); B1 = u.h;
    }
    a00 = wmma_f16(a00, A0, B0); a01 = wmma_f16(a01, A0, B1);
    a10 = wmma_f16(a10, A1, B0); a11 = wmma_f16(a11, A1, B1);
  }
  int nn = lane & 15, mo = (lane >> 4) << 3;
#pragma unroll
  for (int r = 0; r < 8; ++r) {
    int ma = m0 + mo + r, mb = m0 + 16 + mo + r;
    out[(size_t)ma * NPIX + n0 + nn]      = a00[r];
    out[(size_t)ma * NPIX + n0 + 16 + nn] = a01[r];
    out[(size_t)mb * NPIX + n0 + nn]      = a10[r];
    out[(size_t)mb * NPIX + n0 + 16 + nn] = a11[r];
  }
}

// ---------------------------------------------------------------------------
extern "C" void kernel_launch(void* const* d_in, const int* in_sizes, int n_in,
                              void* d_out, int out_size, void* d_ws, size_t ws_size,
                              hipStream_t stream) {
  const float* fg    = (const float*)d_in[0];
  const float* mask  = (const float*)d_in[1];
  const float* ga_wq = (const float*)d_in[2];
  const float* ga_bq = (const float*)d_in[3];
  const float* ga_wk = (const float*)d_in[4];
  const float* ga_bk = (const float*)d_in[5];
  const float* ga_wv = (const float*)d_in[6];
  const float* ga_bv = (const float*)d_in[7];
  const float* ga_g  = (const float*)d_in[8];
  const float* pa_wq = (const float*)d_in[9];
  const float* pa_bq = (const float*)d_in[10];
  const float* pa_wk = (const float*)d_in[11];
  const float* pa_bk = (const float*)d_in[12];
  const float* pa_wv = (const float*)d_in[13];
  const float* pa_bv = (const float*)d_in[14];
  const float* pa_g  = (const float*)d_in[15];
  float* outp = (float*)d_out;
  (void)in_sizes; (void)n_in; (void)out_size; (void)ws_size;

  char* base = (char*)d_ws;
  size_t off = 0;
  auto alloc = [&](size_t bytes) -> void* {
    void* p = base + off;
    off = (off + bytes + 255) & ~(size_t)255;
    return p;
  };

  f16*   w16     = (f16*)  alloc(5 * 4096 * sizeof(f16));
  float* wks     = (float*)alloc(65 * sizeof(float));
  f16*   a16s    = (f16*)  alloc((size_t)CCH * NPIX * 2);
  f16*   bg16s   = (f16*)  alloc((size_t)CCH * NPIX * 2);
  f16*   qT16    = (f16*)  alloc((size_t)NPIX * CCH * 2);
  f16*   k16s    = (f16*)  alloc((size_t)CCH * NPIX * 2);
  f16*   vT16s   = (f16*)  alloc((size_t)NPIX * CCH * 2);
  float* sim     = (float*)alloc((size_t)NPIX * NPIX * 4);   // aliased as cr
  f16*   attn16  = (f16*)  alloc((size_t)NPIX * NPIX * 2);   // aliased as attn2s
  float* fgout   = (float*)alloc((size_t)CCH * NPIX * 4);
  f16*   xu16s   = (f16*)  alloc((size_t)CCH * LP * 2);
  float* mu      = (float*)alloc((size_t)LP * 4);
  float* qpaT    = (float*)alloc((size_t)LP * CCH * 4);
  float* vpaT    = (float*)alloc((size_t)LP * CCH * 4);
  float* ksum    = (float*)alloc((size_t)LP * 4);
  f16*   ckA_cr  = (f16*)  alloc((size_t)NPIX * KCR * 2);
  f16*   ckA_rec = (f16*)  alloc((size_t)CCH * LP * 2);
  float* mm      = (float*)alloc((size_t)NPIX * 4);
  f16*   Bcrs    = (f16*)  alloc((size_t)KCR * NPIX * 2);
  float* cr2     = (float*)alloc((size_t)NPIX * NPIX * 4);
  float* cr      = sim;        // reuse (global stage done before local stage)
  f16*   attn2s  = attn16;     // reuse

  k_prep_weights<<<80, 256, 0, stream>>>(ga_wq, ga_wk, ga_wv, pa_wq, pa_wv,
                                         pa_wk, pa_bk, w16, wks);

  for (int b = 0; b < 2; ++b) {
    const float* fgb = fg + (size_t)b * CCH * NPIX;
    const float* mb  = mask + (size_t)b * NPIX;
    float* outb = outp + (size_t)b * CCH * NPIX;

    // ---- global attention ----
    k_pack_fg<<<1024, 256, 0, stream>>>(fgb, mb, a16s, bg16s);
    k_gemm_qkv<1><<<dim3(256, 4), 32, 0, stream>>>(w16,        a16s,  ga_bq, qT16);
    k_gemm_qkv<0><<<dim3(256, 4), 32, 0, stream>>>(w16 + 4096, bg16s, ga_bk, k16s);
    k_gemm_qkv<2><<<dim3(256, 4), 32, 0, stream>>>(w16 + 8192, a16s,  ga_bv, vT16s);
    k_gemm_sim<<<dim3(128, 128), 32, 0, stream>>>(qT16, k16s, mb, sim);
    k_softmax_row<<<4096, 256, 0, stream>>>(sim, attn16);
    k_gemm_out<<<dim3(2, 128), 32, 0, stream>>>(attn16, vT16s, fgb, mb, ga_g, fgout);

    // ---- patch attention ----
    k_pack_xu<<<9216, 256, 0, stream>>>(fgb, mb, xu16s);
    k_pack_mu<<<144, 256, 0, stream>>>(mb, mu);
    k_gemm_pa<<<dim3(2304, 4), 32, 0, stream>>>(w16 + 12288, xu16s, pa_bq, qpaT);
    k_gemm_pa<<<dim3(2304, 4), 32, 0, stream>>>(w16 + 16384, xu16s, pa_bv, vpaT);
    k_ksum<<<144, 256, 0, stream>>>(xu16s, wks, ksum);
    k_patch<<<4096, 64, 0, stream>>>(xu16s, qpaT, vpaT, ksum, mu, pa_g,
                                     ckA_cr, ckA_rec, mm);

    // ---- dynamic conv + propagation + softmax + reconstruction ----
    k_pack_bcr<<<9216, 256, 0, stream>>>(fgout, Bcrs);
    k_gemm_cr<<<dim3(128, 128), 32, 0, stream>>>(ckA_cr, Bcrs, cr);
    k_box<<<4096, 256, 0, stream>>>(cr, cr2);
    k_colsoftmax<<<64, 256, 0, stream>>>(cr2, mm, attn2s);
    k_gemm_rec<<<dim3(128, 2), 32, 0, stream>>>(ckA_rec, attn2s, outb);
  }
}
